// LSTM_20358144983305
// MI455X (gfx1250) — matrix operand order repaired and checked by reading, not compiled, not run on hardware
//
#include <hip/hip_runtime.h>

// ---------------------------------------------------------------------------
// LSTM (B=64, T=512, I=512, H=1024, O=256) for gfx1250 (MI455X).
//  - bf16 V_WMMA_F32_16X16X32_BF16, f32 accumulation.
//  - Input projection Gx[t] = x_t @ Wx.T + b precomputed for ALL t as one
//    massively parallel WMMA GEMM (recurrence-independent); sequential
//    per-step kernel only does h @ Wh.T (K=1024) + fused pointwise update.
//  - B-tiles double-buffered in LDS via GLOBAL_LOAD_ASYNC_TO_LDS_B128
//    (ASYNCcnt) when the toolchain exposes the builtin.
//  - Weights (12.6 MB bf16) stay L2-resident (192 MB) across all 512 steps.
// ---------------------------------------------------------------------------

typedef __attribute__((ext_vector_type(16))) __bf16 v16bf;
typedef __attribute__((ext_vector_type(8)))  float  v8f;
typedef __attribute__((ext_vector_type(4)))  int    v4i;

#define Bsz  64
#define Tseq 512
#define Isz  512
#define Hsz  1024
#define Osz  256
#define Ktot (Isz + Hsz)   // 1536
#define NG   4             // gates f, i, C, o
#define G4H  (NG * Hsz)    // 4096

#if defined(__has_builtin)
#  if __has_builtin(__builtin_amdgcn_global_load_async_to_lds_b128)
#    define HAVE_ASYNC_LDS 1
#  endif
#endif
#ifndef HAVE_ASYNC_LDS
#  define HAVE_ASYNC_LDS 0
#endif

#if HAVE_ASYNC_LDS
// addrspace(1) == HIP "__device__" (global), addrspace(3) == "__shared__" (LDS)
typedef __attribute__((address_space(1))) v4i* gptr_v4i;
typedef __attribute__((address_space(3))) v4i* lptr_v4i;
#endif

union FragBF {
    v16bf v;
    uint4 q[2];
};

__device__ __forceinline__ unsigned short f2bf(float f) {
    unsigned u = __float_as_uint(f);
    unsigned r = u + 0x7FFFu + ((u >> 16) & 1u);
    return (unsigned short)(r >> 16);
}
__device__ __forceinline__ float sigmoidf(float x) {
    return 1.0f / (1.0f + __expf(-x));
}

__device__ __forceinline__ void wait_async_lds() {
#if HAVE_ASYNC_LDS
#  if __has_builtin(__builtin_amdgcn_s_wait_asynccnt)
    __builtin_amdgcn_s_wait_asynccnt(0);
#  else
    asm volatile("s_wait_asynccnt 0x0" ::: "memory");
#  endif
#endif
}

// Stage the 4 gate B-tiles (16 rows x 32 K, bf16 each) for columns
// [kcol, kcol+32) of weight rows [g*H + j0, g*H + j0 + 16).
// 128 threads x 2 chunks of 16B. Async DMA to LDS when available.
__device__ __forceinline__ void stage_gate_tiles(
    const unsigned short* __restrict__ W,     // [4H][Ktot] bf16 row-major
    int j0, int kcol,
    unsigned short (*dst)[16][32],            // one LDS buffer: [NG][16][32]
    int tid)
{
    int c = tid * 2;
#pragma unroll
    for (int it = 0; it < 2; ++it, ++c) {
        const int gn   = c >> 2;     // (gate, n)
        const int part = c & 3;      // 8 bf16 per 16B chunk
        const int g = gn >> 4, n = gn & 15;
        const unsigned short* src =
            W + (size_t)(g * Hsz + j0 + n) * Ktot + kcol + part * 8;
#if HAVE_ASYNC_LDS
        __builtin_amdgcn_global_load_async_to_lds_b128(
            (gptr_v4i)(void*)src,
            (lptr_v4i)(void*)&dst[g][n][part * 8], 0, 0);
#else
        *(uint4*)&dst[g][n][part * 8] = *(const uint4*)src;
#endif
        __builtin_prefetch((const void*)(src + 32), 0, 1);  // global_prefetch_b8
    }
}

__device__ __forceinline__ v8f wmma_acc(const FragBF& a,
                                        const unsigned short* bsrc, v8f acc) {
    FragBF b;
    b.q[0] = *(const uint4*)bsrc;
    b.q[1] = *(const uint4*)(bsrc + 16);
    return __builtin_amdgcn_wmma_f32_16x16x32_bf16(false, a.v, false, b.v,
                                                   (short)0, acc, false, false);
}

// --------------------------- conversion / init -----------------------------

__global__ void f32_to_bf16_kernel(const float* __restrict__ src,
                                   unsigned short* __restrict__ dst, int n) {
    int i = blockIdx.x * blockDim.x + threadIdx.x;
    int stride = gridDim.x * blockDim.x;
    for (; i < n; i += stride) dst[i] = f2bf(src[i]);
}

__global__ void init_state_kernel(unsigned short* __restrict__ hb,
                                  float* __restrict__ Cst, int n) {
    int i = blockIdx.x * blockDim.x + threadIdx.x;
    int stride = gridDim.x * blockDim.x;
    for (; i < n; i += stride) { hb[i] = 0; Cst[i] = 0.0f; }
}

// --------------------- Gx precompute (parallel GEMM) -----------------------
// Gx[t][b][4H] = x[b,t] @ Wx.T + bias.  M = B*T = 32768 flattened rows
// (r = b*T + t), K = 512 (weight columns [H, H+I)), N = 4096.
// grid = (64 j-tiles, 512 row-groups), block = 128 (4 waves x 16-row M-tile).

__global__ __launch_bounds__(128) void gx_precompute_kernel(
    const unsigned short* __restrict__ Wb,    // [4H][Ktot]
    const unsigned short* __restrict__ xb,    // rows (b*T+t), stride I
    const float* __restrict__ b_f, const float* __restrict__ b_i,
    const float* __restrict__ b_C, const float* __restrict__ b_o,
    float* __restrict__ Gx)                   // [T][B][4H]
{
    __shared__ unsigned short Bs[2][NG][16][32];   // 8 KB double buffer
    const int tid = threadIdx.x, wave = tid >> 5, lane = tid & 31;
    const int half = lane >> 4, ln = lane & 15;
    const int j0 = blockIdx.x * 16;
    const int r0 = blockIdx.y * 64 + wave * 16;

    v8f acc0 = {}, acc1 = {}, acc2 = {}, acc3 = {};

    stage_gate_tiles(Wb, j0, Hsz, Bs[0], tid);
    const int NIT = Isz / 32;
    for (int it = 0; it < NIT; ++it) {
        const int k0  = it * 32;
        const int cur = it & 1;
        wait_async_lds();
        __syncthreads();  // buf[cur] ready; prior reads of buf[1-cur] drained
        if (it + 1 < NIT)
            stage_gate_tiles(Wb, j0, Hsz + k0 + 32, Bs[1 - cur], tid);

        FragBF a;  // A: lanes 0-15 K {0..7,16..23}; lanes 16-31 K {8..15,24..31}
        const unsigned short* ap = xb + (size_t)(r0 + ln) * Isz + k0;
        a.q[0] = *(const uint4*)(ap + half * 8);
        a.q[1] = *(const uint4*)(ap + 16 + half * 8);

        const int koff = half * 8;
        acc0 = wmma_acc(a, &Bs[cur][0][ln][koff], acc0);
        acc1 = wmma_acc(a, &Bs[cur][1][ln][koff], acc1);
        acc2 = wmma_acc(a, &Bs[cur][2][ln][koff], acc2);
        acc3 = wmma_acc(a, &Bs[cur][3][ln][koff], acc3);
    }

    const int j = j0 + ln;
    const float bf_ = b_f[j], bi_ = b_i[j], bC_ = b_C[j], bo_ = b_o[j];
#pragma unroll
    for (int v = 0; v < 8; ++v) {
        const int r = r0 + v + half * 8;
        const int b = r >> 9, t = r & (Tseq - 1);   // r = b*T + t
        float* g = Gx + ((size_t)t * Bsz + b) * G4H + j;
        g[0 * Hsz] = acc0[v] + bf_;
        g[1 * Hsz] = acc1[v] + bi_;
        g[2 * Hsz] = acc2[v] + bC_;
        g[3 * Hsz] = acc3[v] + bo_;
    }
}

// ------------------ fused recurrent step (Gx-precomputed) ------------------
// gates = h_prev @ Wh.T + Gx[t]; K = 1024 only. grid = 64 j-tiles x 128 thr.

__global__ __launch_bounds__(128) void lstm_step_gx_kernel(
    const unsigned short* __restrict__ Wb,     // [4H][Ktot]; cols [0,H)
    const float* __restrict__ Gx,              // [T][B][4H]
    const unsigned short* __restrict__ hprev,  // [B][H] bf16
    unsigned short* __restrict__ hnext,        // [B][H] bf16
    float* __restrict__ Cst,                   // [B][H] f32
    int t)
{
    __shared__ unsigned short Bs[2][NG][16][32];
    const int tid = threadIdx.x, wave = tid >> 5, lane = tid & 31;
    const int half = lane >> 4, ln = lane & 15;
    const int j0 = blockIdx.x * 16, m0 = wave * 16;

    v8f acc0 = {}, acc1 = {}, acc2 = {}, acc3 = {};

    stage_gate_tiles(Wb, j0, 0, Bs[0], tid);
    const int NIT = Hsz / 32;
    for (int it = 0; it < NIT; ++it) {
        const int k0  = it * 32;
        const int cur = it & 1;
        wait_async_lds();
        __syncthreads();
        if (it + 1 < NIT)
            stage_gate_tiles(Wb, j0, k0 + 32, Bs[1 - cur], tid);

        FragBF a;
        const unsigned short* ap = hprev + (size_t)(m0 + ln) * Hsz + k0;
        a.q[0] = *(const uint4*)(ap + half * 8);
        a.q[1] = *(const uint4*)(ap + 16 + half * 8);

        const int koff = half * 8;
        acc0 = wmma_acc(a, &Bs[cur][0][ln][koff], acc0);
        acc1 = wmma_acc(a, &Bs[cur][1][ln][koff], acc1);
        acc2 = wmma_acc(a, &Bs[cur][2][ln][koff], acc2);
        acc3 = wmma_acc(a, &Bs[cur][3][ln][koff], acc3);
    }

    const int j = j0 + ln;
    const float* gx = Gx + (size_t)t * Bsz * G4H;
#pragma unroll
    for (int v = 0; v < 8; ++v) {
        const int m = m0 + v + half * 8;
        const float* g = gx + (size_t)m * G4H + j;
        const size_t idx = (size_t)m * Hsz + j;
        const float fg = sigmoidf(acc0[v] + g[0 * Hsz]);
        const float ig = sigmoidf(acc1[v] + g[1 * Hsz]);
        const float cg = tanhf  (acc2[v] + g[2 * Hsz]);
        const float og = sigmoidf(acc3[v] + g[3 * Hsz]);
        const float Cn = fg * Cst[idx] + ig * cg;
        Cst[idx]   = Cn;
        hnext[idx] = f2bf(og * tanhf(Cn));
    }
}

// ------------- fallback fused step (no Gx; K = 1536 with bias) -------------

__global__ __launch_bounds__(128) void lstm_step_kernel(
    const unsigned short* __restrict__ Wb,
    const unsigned short* __restrict__ xb,
    const unsigned short* __restrict__ hprev,
    unsigned short* __restrict__ hnext,
    float* __restrict__ Cst,
    const float* __restrict__ b_f, const float* __restrict__ b_i,
    const float* __restrict__ b_C, const float* __restrict__ b_o,
    int t)
{
    __shared__ unsigned short Bs[2][NG][16][32];
    const int tid = threadIdx.x, wave = tid >> 5, lane = tid & 31;
    const int half = lane >> 4, ln = lane & 15;
    const int j0 = blockIdx.x * 16, m0 = wave * 16;

    v8f acc0 = {}, acc1 = {}, acc2 = {}, acc3 = {};

    stage_gate_tiles(Wb, j0, 0, Bs[0], tid);
    const int NIT = Ktot / 32;
    for (int it = 0; it < NIT; ++it) {
        const int k0  = it * 32;
        const int cur = it & 1;
        wait_async_lds();
        __syncthreads();
        if (it + 1 < NIT)
            stage_gate_tiles(Wb, j0, k0 + 32, Bs[1 - cur], tid);

        FragBF a;
        {
            const int m = m0 + ln;
            const unsigned short* ap;
            if (k0 < Hsz)  // uniform (scalar) branch
                ap = hprev + (size_t)m * Hsz + k0;
            else
                ap = xb + ((size_t)m * Tseq + t) * Isz + (k0 - Hsz);
            a.q[0] = *(const uint4*)(ap + half * 8);
            a.q[1] = *(const uint4*)(ap + 16 + half * 8);
        }

        const int koff = half * 8;
        acc0 = wmma_acc(a, &Bs[cur][0][ln][koff], acc0);
        acc1 = wmma_acc(a, &Bs[cur][1][ln][koff], acc1);
        acc2 = wmma_acc(a, &Bs[cur][2][ln][koff], acc2);
        acc3 = wmma_acc(a, &Bs[cur][3][ln][koff], acc3);
    }

    const int j = j0 + ln;
    const float bf_ = b_f[j], bi_ = b_i[j], bC_ = b_C[j], bo_ = b_o[j];
#pragma unroll
    for (int v = 0; v < 8; ++v) {
        const int m = m0 + v + half * 8;
        const size_t idx = (size_t)m * Hsz + j;
        const float fg = sigmoidf(acc0[v] + bf_);
        const float ig = sigmoidf(acc1[v] + bi_);
        const float cg = tanhf  (acc2[v] + bC_);
        const float og = sigmoidf(acc3[v] + bo_);
        const float Cn = fg * Cst[idx] + ig * cg;
        Cst[idx]   = Cn;
        hnext[idx] = f2bf(og * tanhf(Cn));
    }
}

// ------------------------------ final FC -----------------------------------

__global__ __launch_bounds__(32) void fc_kernel(
    const unsigned short* __restrict__ hb,    // [B][H] bf16 (final h)
    const unsigned short* __restrict__ fcwb,  // [O][H] bf16
    const float* __restrict__ fc_b,
    float* __restrict__ out)                  // [B][O] f32
{
    const int lane = threadIdx.x & 31;
    const int half = lane >> 4, ln = lane & 15;
    const int m0 = (blockIdx.x & 3) * 16;
    const int n0 = (blockIdx.x >> 2) * 16;

    v8f acc = {};
    for (int k0 = 0; k0 < Hsz; k0 += 32) {
        FragBF a, b;
        const unsigned short* ap = hb + (size_t)(m0 + ln) * Hsz + k0;
        a.q[0] = *(const uint4*)(ap + half * 8);
        a.q[1] = *(const uint4*)(ap + 16 + half * 8);
        const unsigned short* bp = fcwb + (size_t)(n0 + ln) * Hsz + k0 + half * 8;
        b.q[0] = *(const uint4*)bp;
        b.q[1] = *(const uint4*)(bp + 16);
        acc = __builtin_amdgcn_wmma_f32_16x16x32_bf16(false, a.v, false, b.v,
                                                      (short)0, acc, false, false);
    }
    const int o = n0 + ln;
    const float bias = fc_b[o];
#pragma unroll
    for (int v = 0; v < 8; ++v) {
        const int m = m0 + v + half * 8;
        out[(size_t)m * Osz + o] = acc[v] + bias;
    }
}

// ------------------------------- launcher ----------------------------------

extern "C" void kernel_launch(void* const* d_in, const int* in_sizes, int n_in,
                              void* d_out, int out_size, void* d_ws, size_t ws_size,
                              hipStream_t stream) {
    (void)in_sizes; (void)n_in; (void)out_size;

    const float* x    = (const float*)d_in[0];
    const float* W_f  = (const float*)d_in[1];
    const float* W_i  = (const float*)d_in[2];
    const float* W_C  = (const float*)d_in[3];
    const float* W_o  = (const float*)d_in[4];
    const float* b_f  = (const float*)d_in[5];
    const float* b_i  = (const float*)d_in[6];
    const float* b_C  = (const float*)d_in[7];
    const float* b_o  = (const float*)d_in[8];
    const float* fc_w = (const float*)d_in[9];
    const float* fc_b = (const float*)d_in[10];

    char* ws = (char*)d_ws;
    size_t off = 0;
    unsigned short* Wb   = (unsigned short*)(ws + off); off += (size_t)G4H * Ktot * 2;
    unsigned short* fcwb = (unsigned short*)(ws + off); off += (size_t)Osz * Hsz * 2;
    unsigned short* xb   = (unsigned short*)(ws + off); off += (size_t)Bsz * Tseq * Isz * 2;
    unsigned short* hb0  = (unsigned short*)(ws + off); off += (size_t)Bsz * Hsz * 2;
    unsigned short* hb1  = (unsigned short*)(ws + off); off += (size_t)Bsz * Hsz * 2;
    float*          Cst  = (float*)(ws + off);          off += (size_t)Bsz * Hsz * 4;
    float*          Gx   = (float*)(ws + off);
    const size_t need_gx = off + (size_t)Tseq * Bsz * G4H * 4;  // +512 MB
    const bool use_gx = (ws_size >= need_gx);

    // one-time conversions (gate order f,i,C,o matches reference concat)
    const int nWg = Hsz * Ktot;
    f32_to_bf16_kernel<<<1024, 256, 0, stream>>>(W_f, Wb + 0 * (size_t)nWg, nWg);
    f32_to_bf16_kernel<<<1024, 256, 0, stream>>>(W_i, Wb + 1 * (size_t)nWg, nWg);
    f32_to_bf16_kernel<<<1024, 256, 0, stream>>>(W_C, Wb + 2 * (size_t)nWg, nWg);
    f32_to_bf16_kernel<<<1024, 256, 0, stream>>>(W_o, Wb + 3 * (size_t)nWg, nWg);
    f32_to_bf16_kernel<<<1024, 256, 0, stream>>>(fc_w, fcwb, Osz * Hsz);
    f32_to_bf16_kernel<<<2048, 256, 0, stream>>>(x, xb, Bsz * Tseq * Isz);
    init_state_kernel<<<256, 256, 0, stream>>>(hb0, Cst, Bsz * Hsz);

    if (use_gx) {
        // recurrence-independent input projection for all timesteps at once
        gx_precompute_kernel<<<dim3(Hsz / 16, (Bsz * Tseq) / 64), 128, 0, stream>>>(
            Wb, xb, b_f, b_i, b_C, b_o, Gx);
    }

    unsigned short* hprev = hb0;
    unsigned short* hnext = hb1;
    for (int t = 0; t < Tseq; ++t) {
        if (use_gx) {
            lstm_step_gx_kernel<<<Hsz / 16, 128, 0, stream>>>(
                Wb, Gx, hprev, hnext, Cst, t);
        } else {
            lstm_step_kernel<<<Hsz / 16, 128, 0, stream>>>(
                Wb, xb, hprev, hnext, Cst, b_f, b_i, b_C, b_o, t);
        }
        unsigned short* tmp = hprev; hprev = hnext; hnext = tmp;
    }

    fc_kernel<<<64, 32, 0, stream>>>(hprev, fcwb, fc_b, (float*)d_out);
}
